// HPT_TAGConvNet_14388140441685
// MI455X (gfx1250) — matrix operand order repaired and checked by reading, not compiled, not run on hardware
//
#include <hip/hip_runtime.h>

// ---------------------------------------------------------------------------
// TAGConvNet for MI455X (gfx1250, wave32, WMMA)
// N=100000 nodes, E=600000 edges, F=6, C=128, K=3 hops, 2 TAG layers, 2 MLP
// ---------------------------------------------------------------------------

#define NNODES 100000
#define NEDGES 600000
#define FIN    6
#define CH     128
#define KHOP   3

typedef __attribute__((ext_vector_type(16))) __bf16        v16bf;
typedef __attribute__((ext_vector_type(8)))  float         v8f;
typedef __attribute__((ext_vector_type(8)))  unsigned int  v8u;

// float -> bf16 with round-to-nearest-even (bit-exact RNE truncation)
static __device__ __forceinline__ unsigned int bf16bits(float f) {
    unsigned int u = __builtin_bit_cast(unsigned int, f);
    u += ((u >> 16) & 1u) + 0x7FFFu;
    return u >> 16;
}
static __device__ __forceinline__ unsigned int pack2bf(float a, float b) {
    return bf16bits(a) | (bf16bits(b) << 16);
}

// ---------------------------------------------------------------------------
// Utility kernels
// ---------------------------------------------------------------------------
__global__ void zero_kernel(float* __restrict__ p, int n) {
    int i = blockIdx.x * blockDim.x + threadIdx.x;
    if (i < n) p[i] = 0.0f;
}

__global__ void deg_kernel(const int* __restrict__ coli, float* __restrict__ deg, int E) {
    int e = blockIdx.x * blockDim.x + threadIdx.x;
    if (e < E) atomicAdd(&deg[coli[e]], 1.0f);
}

__global__ void dis_kernel(float* __restrict__ deg, int n) {
    int i = blockIdx.x * blockDim.x + threadIdx.x;
    if (i < n) {
        float d = deg[i];
        deg[i] = (d > 0.0f) ? rsqrtf(fmaxf(d, 1.0f)) : 0.0f;
    }
}

__global__ void norm_kernel(const int* __restrict__ rowi, const int* __restrict__ coli,
                            const float* __restrict__ dis, float* __restrict__ norm, int E) {
    int e = blockIdx.x * blockDim.x + threadIdx.x;
    if (e < E) norm[e] = dis[rowi[e]] * dis[coli[e]];
}

// h = relu(x @ W0 + b0) ; x:[N,6], W0:[6,128]
__global__ void lin0_kernel(const float* __restrict__ x, const float* __restrict__ W0,
                            const float* __restrict__ b0, float* __restrict__ h, int n) {
    int idx = blockIdx.x * blockDim.x + threadIdx.x;
    if (idx >= n * CH) return;
    int node = idx >> 7, ch = idx & (CH - 1);
    const float* xp = x + node * FIN;
    float s = b0[ch];
#pragma unroll
    for (int f = 0; f < FIN; ++f) s += xp[f] * W0[f * CH + ch];
    h[idx] = s > 0.0f ? s : 0.0f;
}

// One wave per edge: gather float4 of source row, scale by norm, atomic scatter.
__global__ void spmm_edge(const int* __restrict__ rowi, const int* __restrict__ coli,
                          const float* __restrict__ norm, const float* __restrict__ x,
                          float* __restrict__ y, int E) {
    int e = blockIdx.x * (blockDim.x >> 5) + (threadIdx.x >> 5);
    if (e >= E) return;
    int lane = threadIdx.x & 31;
    int r = rowi[e], c = coli[e];
    float nv = norm[e];
    const float4 v = ((const float4*)(x + (size_t)r * CH))[lane];
    float* yp = y + (size_t)c * CH + lane * 4;
    atomicAdd(yp + 0, nv * v.x);
    atomicAdd(yp + 1, nv * v.y);
    atomicAdd(yp + 2, nv * v.z);
    atomicAdd(yp + 3, nv * v.w);
}

// out[i] = relu(dot(h[i,:], W1) + b1)
__global__ void final_kernel(const float* __restrict__ h, const float* __restrict__ W1,
                             const float* __restrict__ b1, float* __restrict__ out, int n) {
    int i = blockIdx.x * blockDim.x + threadIdx.x;
    if (i >= n) return;
    const float* hp = h + (size_t)i * CH;
    float s = b1[0];
#pragma unroll 8
    for (int c = 0; c < CH; ++c) s += hp[c] * W1[c];
    out[i] = s > 0.0f ? s : 0.0f;
}

// ---------------------------------------------------------------------------
// WMMA GEMM: Cout[M,128] = (Cin? Cin:0) + A[M,128] @ W[128,128]  (+bias, relu)
// Block = 128 threads = 4 waves.
// W staged in LDS in *fragment-ready* order: for each (k-step ks, col-tile t),
// lane L's 8 packed-bf16 dwords are contiguous -> two ds_load_b128 per
// B fragment, no per-element shuffling into the WMMA operand octet.
// Each wave computes a 16x128 output tile (8 col tiles x 4 K-steps = 32 WMMAs).
// ---------------------------------------------------------------------------
__global__ __launch_bounds__(128) void tag_gemm128(
    const float* __restrict__ A,
    const float* __restrict__ W,
    const float* __restrict__ Cin,
    float* __restrict__ Cout,
    const float* __restrict__ bias,
    int M, int accumulate, int relu)
{
    // Layout: wlds[(((ks*8 + t)*32 + lane)*8) + j]
    //   n  = t*16 + (lane & 15)
    //   k0 = ks*32 + ((lane>>4)<<3) + ((j>>2)<<4) + 2*(j&3)      (B fragment per ISA 7.12.2)
    //   entry = { bf16(W[k0][n]), bf16(W[k0+1][n]) } packed in one dword
    __shared__ unsigned int wlds[8192];  // 32 KB
    for (int e = threadIdx.x; e < 8192; e += blockDim.x) {
        int j  = e & 7;
        int ln = (e >> 3) & 31;
        int t  = (e >> 8) & 7;
        int ks = e >> 11;
        int n  = t * 16 + (ln & 15);
        int k0 = ks * 32 + ((ln >> 4) << 3) + ((j >> 2) << 4) + 2 * (j & 3);
        wlds[e] = pack2bf(W[k0 * CH + n], W[(k0 + 1) * CH + n]);
    }
    __syncthreads();

    const int wave = threadIdx.x >> 5;
    const int lane = threadIdx.x & 31;
    const int m0 = (blockIdx.x * 4 + wave) * 16;
    if (m0 >= M) return;  // uniform per wave -> EXEC stays all ones for WMMA

    // A-fragment lane mapping (16-bit A 16x32, ISA 7.12.2):
    //   lanes 0-15: row M=lane, K in {0..7} U {16..23}; lanes 16-31: K+8
    const int mA = m0 + (lane & 15);
    const int kA = (lane >> 4) << 3;           // 0 or 8
    // C/D fragment: VGPR r -> (M = r + 8*(lane>=16), N = lane&15)
    const int mC = m0 + ((lane >> 4) << 3);
    const int nC = lane & 15;

    v8f c[8];
#pragma unroll
    for (int t = 0; t < 8; ++t) {
        if (accumulate) {
#pragma unroll
            for (int r = 0; r < 8; ++r)
                c[t][r] = Cin[(size_t)(mC + r) * CH + t * 16 + nC];
        } else {
#pragma unroll
            for (int r = 0; r < 8; ++r) c[t][r] = 0.0f;
        }
    }

#pragma unroll
    for (int ks = 0; ks < 4; ++ks) {
        const int kk = ks * 32;
        // A fragment: two contiguous 8-float runs -> 4x float4 loads,
        // packed in-register to bf16 pairs.
        const float* ap = A + (size_t)mA * CH + kk + kA;  // 16B aligned (kA in {0,8})
        float4 x0 = ((const float4*)ap)[0];               // K = kA+0..3
        float4 x1 = ((const float4*)ap)[1];               // K = kA+4..7
        float4 x2 = ((const float4*)(ap + 16))[0];        // K = 16+kA+0..3
        float4 x3 = ((const float4*)(ap + 16))[1];        // K = 16+kA+4..7
        v8u au;
        au[0] = pack2bf(x0.x, x0.y);
        au[1] = pack2bf(x0.z, x0.w);
        au[2] = pack2bf(x1.x, x1.y);
        au[3] = pack2bf(x1.z, x1.w);
        au[4] = pack2bf(x2.x, x2.y);
        au[5] = pack2bf(x2.z, x2.w);
        au[6] = pack2bf(x3.x, x3.y);
        au[7] = pack2bf(x3.z, x3.w);
        const v16bf a = __builtin_bit_cast(v16bf, au);

        const unsigned int* wbase = wlds + ((ks * 8) * 32 + lane) * 8;
#pragma unroll
        for (int t = 0; t < 8; ++t) {
            // 32 contiguous bytes per lane -> two ds_load_b128
            v8u bu = *(const v8u*)(wbase + t * 32 * 8);
            const v16bf b = __builtin_bit_cast(v16bf, bu);
            c[t] = __builtin_amdgcn_wmma_f32_16x16x32_bf16(
                false, a, false, b, (short)0, c[t], false, false);
        }
    }

#pragma unroll
    for (int t = 0; t < 8; ++t) {
#pragma unroll
        for (int r = 0; r < 8; ++r) {
            float v = c[t][r];
            if (bias) v += bias[t * 16 + nC];
            if (relu) v = v > 0.0f ? v : 0.0f;
            Cout[(size_t)(mC + r) * CH + t * 16 + nC] = v;
        }
    }
}

// ---------------------------------------------------------------------------
// Launch
// ---------------------------------------------------------------------------
extern "C" void kernel_launch(void* const* d_in, const int* in_sizes, int n_in,
                              void* d_out, int out_size, void* d_ws, size_t ws_size,
                              hipStream_t stream) {
    const float* x     = (const float*)d_in[0];
    const int*   ei    = (const int*)d_in[1];
    const float* W0    = (const float*)d_in[2];
    const float* b0    = (const float*)d_in[3];
    const float* W_tag = (const float*)d_in[4];   // [2, 4, 128, 128]
    const float* b_tag = (const float*)d_in[5];   // [2, 128]
    const float* W_mlp = (const float*)d_in[6];   // [2, 128, 128]
    const float* b_mlp = (const float*)d_in[7];   // [2, 128]
    const float* W1    = (const float*)d_in[8];   // [128]
    const float* b1    = (const float*)d_in[9];   // [1]
    float* out = (float*)d_out;
    (void)in_sizes; (void)n_in; (void)out_size; (void)ws_size;

    const int N = NNODES, E = NEDGES;
    const int* rowi = ei;          // sources
    const int* coli = ei + E;      // targets

    // Workspace carve-out
    char* ws = (char*)d_ws;
    size_t o = 0;
    auto take = [&](size_t bytes) { void* p = ws + o; o += (bytes + 255) & ~(size_t)255; return p; };
    float* deg  = (float*)take((size_t)N * sizeof(float));
    float* norm = (float*)take((size_t)E * sizeof(float));
    float* h    = (float*)take((size_t)N * CH * sizeof(float));
    float* acc  = (float*)take((size_t)N * CH * sizeof(float));
    float* tA   = (float*)take((size_t)N * CH * sizeof(float));
    float* tB   = (float*)take((size_t)N * CH * sizeof(float));

    const int NC = N * CH;

    // --- normalization coefficients ---
    zero_kernel<<<(N + 255) / 256, 256, 0, stream>>>(deg, N);
    deg_kernel<<<(E + 255) / 256, 256, 0, stream>>>(coli, deg, E);
    dis_kernel<<<(N + 255) / 256, 256, 0, stream>>>(deg, N);
    norm_kernel<<<(E + 255) / 256, 256, 0, stream>>>(rowi, coli, deg, norm, E);

    // --- lin0 ---
    lin0_kernel<<<(NC + 255) / 256, 256, 0, stream>>>(x, W0, b0, h, N);

    dim3 gemmGrid((N + 63) / 64);

    // --- TAGConv layers ---
    for (int g = 0; g < 2; ++g) {
        const float* Wg = W_tag + (size_t)g * (KHOP + 1) * CH * CH;
        // acc = h @ W[g,0]
        tag_gemm128<<<gemmGrid, 128, 0, stream>>>(h, Wg, nullptr, acc, nullptr, N, 0, 0);
        const float* src = h;
        for (int k = 1; k <= KHOP; ++k) {
            float* dst = (k & 1) ? tA : tB;
            zero_kernel<<<(NC + 255) / 256, 256, 0, stream>>>(dst, NC);
            spmm_edge<<<(E + 7) / 8, 256, 0, stream>>>(rowi, coli, norm, src, dst, E);
            int last = (k == KHOP);
            // acc += dst @ W[g,k]; on last hop fuse bias+relu and write back to h
            tag_gemm128<<<gemmGrid, 128, 0, stream>>>(
                dst, Wg + (size_t)k * CH * CH, acc,
                last ? h : acc, last ? (b_tag + g * CH) : nullptr, N, 1, last);
            src = dst;
        }
    }

    // --- MLP ---
    float* cur = h;
    float* nxt = acc;
    for (int m = 0; m < 2; ++m) {
        tag_gemm128<<<gemmGrid, 128, 0, stream>>>(
            cur, W_mlp + (size_t)m * CH * CH, nullptr, nxt, b_mlp + m * CH, N, 0, 1);
        float* t = cur; cur = nxt; nxt = t;
    }

    // --- final projection ---
    final_kernel<<<(N + 255) / 256, 256, 0, stream>>>(cur, W1, b1, out, N);
}
